// MultiHeadSelfAttention_83382495084647
// MI455X (gfx1250) — matrix operand order repaired and checked by reading, not compiled, not run on hardware
//
#include <hip/hip_runtime.h>
#include <hip/hip_bf16.h>

// ---------------------------------------------------------------------------
// Multi-head self-attention for MI455X (gfx1250), wave32 + WMMA f16->f32.
//   E=1024, H=16, Dh=64, B=4, L=2048, M = B*L = 8192
// Pipeline: cvt weights/activations -> f16 once; QKV GEMMs (async-LDS staged
// weights, double-buffered); flash attention; output GEMM -> f32.
// ---------------------------------------------------------------------------

typedef _Float16 half8 __attribute__((ext_vector_type(8)));
typedef _Float16 v16h  __attribute__((ext_vector_type(16)));
typedef float    v8f   __attribute__((ext_vector_type(8)));

#define EMBED 1024
#define NHEAD 16
#define HDIM  64
#define BATCH 4
#define SEQ   2048
#define MTOT  (BATCH*SEQ)

__device__ __forceinline__ v16h cat16(half8 lo, half8 hi) {
  return __builtin_shufflevector(lo, hi, 0,1,2,3,4,5,6,7,8,9,10,11,12,13,14,15);
}

// A-matrix fragment 16x32 f16 (M x K). Per ISA 7.12.2:
//   lane l: row m = l&15, g = l>>4; halves 0-7 -> K = 8g..8g+7,
//   halves 8-15 -> K = 16+8g..23+8g.  Two 16B loads per lane.
__device__ __forceinline__ v16h ldA_h(const _Float16* __restrict__ base, int stride, int lane) {
  const _Float16* p = base + (size_t)(lane & 15) * stride + 8 * (lane >> 4);
  return cat16(*(const half8*)p, *(const half8*)(p + 16));
}
// B-matrix fragment 32x16 f16 (K x N), K-major per output column:
//   lane l: column n = l&15, g = l>>4; halves 0-7 -> K = 8g..8g+7,
//   halves 8-15 -> K = 16+8g..23+8g (same as A).
__device__ __forceinline__ v16h ldB_h(const _Float16* __restrict__ base, int stride, int lane) {
  const _Float16* p = base + (size_t)(lane & 15) * stride + 8 * (lane >> 4);
  return cat16(*(const half8*)p, *(const half8*)(p + 16));
}

#define WMMA(a, b, c) \
  __builtin_amdgcn_wmma_f32_16x16x32_f16(false, (a), false, (b), (short)0, (c), false, false)

__device__ __forceinline__ v8f zero8() {
  v8f z = {0.f, 0.f, 0.f, 0.f, 0.f, 0.f, 0.f, 0.f};
  return z;
}

// gfx1250 async copy global->LDS, 16B per lane, tracked by ASYNCcnt.
__device__ __forceinline__ void stage_async_b128(const _Float16* g, const _Float16* lds) {
  unsigned l32 = (unsigned)(uintptr_t)lds;  // LDS aperture: low 32 bits = LDS byte offset
  asm volatile("global_load_async_to_lds_b128 %0, %1, off"
               :: "v"(l32), "v"(g) : "memory");
}
__device__ __forceinline__ void wait_async0() {
  asm volatile("s_wait_asynccnt 0x0" ::: "memory");
}

// ---------------------------------------------------------------------------
// Streaming f32 -> f16 convert (n multiple of 8).
// ---------------------------------------------------------------------------
__global__ void cvt_f32_f16(const float* __restrict__ src, _Float16* __restrict__ dst,
                            int n8) {
  const int i = blockIdx.x * blockDim.x + threadIdx.x;
  if (i >= n8) return;
  const float4* s = (const float4*)(src + (size_t)i * 8);
  float4 a = s[0], b = s[1];
  half8 h;
  h[0] = (_Float16)a.x; h[1] = (_Float16)a.y; h[2] = (_Float16)a.z; h[3] = (_Float16)a.w;
  h[4] = (_Float16)b.x; h[5] = (_Float16)b.y; h[6] = (_Float16)b.z; h[7] = (_Float16)b.w;
  *(half8*)(dst + (size_t)i * 8) = h;
}

// ---------------------------------------------------------------------------
// GEMM: C[m,n] = sum_k A[m,k] * W[n,k] + bias[n]   (A: Mx1024 f16, W: 1024x1024 f16)
// mode 0: out f16 [(b*16+h)*2048 + l]*64 + d       (Q, K  : [B,H,L,D])
// mode 1: out f16 [(b*16+h)*64 + d]*2048 + l       (V     : [B,H,D,L])
// mode 2: out f32 m*1024 + n                       (final : [B,L,E])
// Block: 256 thr = 8 waves; tile 256(M) x 64(N); wave = 32x64 = 8 WMMAs/k-step.
// Weight k-tile (64x32 f16 = 4KB) staged to LDS via async DMA, double-buffered:
// DMA for tile k+1 overlaps WMMAs of tile k.
// ---------------------------------------------------------------------------
__global__ void mha_gemm_wmma(const _Float16* __restrict__ A,
                              const _Float16* __restrict__ W,
                              const float* __restrict__ bias,
                              _Float16* __restrict__ outH,
                              float* __restrict__ outF,
                              int mode) {
  __shared__ _Float16 Wt[2][64 * 32];   // double-buffered weight tile

  const int wave = threadIdx.x >> 5;
  const int lane = threadIdx.x & 31;
  const int m0 = blockIdx.x * 256 + wave * 32;
  const int n0 = blockIdx.y * 64;

  // Async staging map: wave w moves rows 8w..8w+7; lane l -> row 8w+(l>>2),
  // 16-byte chunk (l&3) within the 32-half row.
  const int srow = 8 * wave + (lane >> 2);
  const int schunk = (lane & 3) * 8;
  const _Float16* wsrc = W + (size_t)(n0 + srow) * EMBED + schunk;

  v8f acc[2][4];
#pragma unroll
  for (int i = 0; i < 2; ++i)
#pragma unroll
    for (int j = 0; j < 4; ++j) acc[i][j] = zero8();

  stage_async_b128(wsrc, &Wt[0][srow * 32 + schunk]);

  int cur = 0;
  for (int k = 0; k < EMBED; k += 32) {
    wait_async0();        // my piece of tile `cur` is in LDS
    __syncthreads();      // everyone's piece landed; last iter's reads done
    if (k + 32 < EMBED)
      stage_async_b128(wsrc + k + 32, &Wt[cur ^ 1][srow * 32 + schunk]);

    // Prefetch next A k-tile (global_prefetch_b8).
    if (k + 32 < EMBED)
      __builtin_prefetch(A + (size_t)(m0 + (lane & 15)) * EMBED + k + 32, 0, 1);

    const v16h af0 = ldA_h(A + (size_t)m0 * EMBED + k, EMBED, lane);
    const v16h af1 = ldA_h(A + (size_t)(m0 + 16) * EMBED + k, EMBED, lane);
#pragma unroll
    for (int j = 0; j < 4; ++j) {
      const v16h bf = ldB_h(&Wt[cur][j * 16 * 32], 32, lane);
      acc[0][j] = WMMA(af0, bf, acc[0][j]);
      acc[1][j] = WMMA(af1, bf, acc[1][j]);
    }
    cur ^= 1;
  }

  const int col = lane & 15;
  const int hsel = lane >> 4;
#pragma unroll
  for (int i = 0; i < 2; ++i) {
#pragma unroll
    for (int j = 0; j < 4; ++j) {
#pragma unroll
      for (int r = 0; r < 8; ++r) {
        const int m = m0 + i * 16 + r + 8 * hsel;
        const int n = n0 + j * 16 + col;
        const float v = acc[i][j][r] + bias[n];
        if (mode == 2) {
          outF[(size_t)m * EMBED + n] = v;
        } else {
          const int b = m >> 11, l = m & 2047;
          const int h = n >> 6, d = n & 63;
          if (mode == 0)
            outH[((size_t)(b * NHEAD + h) * SEQ + l) * HDIM + d] = (_Float16)v;
          else
            outH[((size_t)(b * NHEAD + h) * HDIM + d) * SEQ + l] = (_Float16)v;
        }
      }
    }
  }
}

// ---------------------------------------------------------------------------
// Flash attention core. Grid: (L/64, B*H). Block: 128 threads = 4 waves.
// Wave w owns 16 query rows; kv processed in tiles of 32.
//   Qh, Kh : [B,H,L,64] f16;  Vt : [B,H,64,L] f16;  Ah : [B,L,1024] f16
// ---------------------------------------------------------------------------
__global__ void mha_attn_wmma(const _Float16* __restrict__ Qh,
                              const _Float16* __restrict__ Kh,
                              const _Float16* __restrict__ Vt,
                              _Float16* __restrict__ Ah) {
  __shared__ _Float16 Pl[4 * 16 * 32];   // per-wave 16x32 f16 prob tile

  const int wave = threadIdx.x >> 5;
  const int lane = threadIdx.x & 31;
  const int bh = blockIdx.y;                 // b*16 + h
  const int q0 = blockIdx.x * 64 + wave * 16;

  const _Float16* Qb = Qh + ((size_t)bh * SEQ + q0) * HDIM;
  const v16h qf0 = ldA_h(Qb + 0, HDIM, lane);
  const v16h qf1 = ldA_h(Qb + 32, HDIM, lane);

  float mrow[8], lrow[8], alpha[8];
  v8f o[4];
#pragma unroll
  for (int r = 0; r < 8; ++r) { mrow[r] = -1e30f; lrow[r] = 0.f; }
#pragma unroll
  for (int dc = 0; dc < 4; ++dc) o[dc] = zero8();

  _Float16* myP = Pl + wave * (16 * 32);
  const int prow = 8 * (lane >> 4);
  const int pc = lane & 15;

  for (int kv0 = 0; kv0 < SEQ; kv0 += 32) {
    const _Float16* Kb = Kh + ((size_t)bh * SEQ + kv0) * HDIM;
    const v16h kb00 = ldB_h(Kb, HDIM, lane);                 // kv 0-15,  d 0-31
    const v16h kb01 = ldB_h(Kb + 32, HDIM, lane);            // kv 0-15,  d 32-63
    const v16h kb10 = ldB_h(Kb + 16 * HDIM, HDIM, lane);     // kv 16-31, d 0-31
    const v16h kb11 = ldB_h(Kb + 16 * HDIM + 32, HDIM, lane);
    v16h vb[4];
#pragma unroll
    for (int dc = 0; dc < 4; ++dc)
      vb[dc] = ldB_h(Vt + ((size_t)bh * HDIM + dc * 16) * SEQ + kv0, SEQ, lane);

    v8f s0 = WMMA(qf0, kb00, zero8()); s0 = WMMA(qf1, kb01, s0);
    v8f s1 = WMMA(qf0, kb10, zero8()); s1 = WMMA(qf1, kb11, s1);

    // Online softmax. C-layout: row m = r + 8*(lane>>4) spans 16 lanes of one reg.
#pragma unroll
    for (int r = 0; r < 8; ++r) {
      const float a = s0[r] * 0.125f;          // 1/sqrt(64)
      const float b = s1[r] * 0.125f;
      float mx = fmaxf(a, b);
#pragma unroll
      for (int off = 1; off < 16; off <<= 1) mx = fmaxf(mx, __shfl_xor(mx, off, 32));
      const float mnew = fmaxf(mrow[r], mx);
      const float p0 = __expf(a - mnew);
      const float p1 = __expf(b - mnew);
      float rs = p0 + p1;
#pragma unroll
      for (int off = 1; off < 16; off <<= 1) rs += __shfl_xor(rs, off, 32);
      alpha[r] = __expf(mrow[r] - mnew);
      lrow[r] = lrow[r] * alpha[r] + rs;
      mrow[r] = mnew;
      s0[r] = p0;
      s1[r] = p1;
    }
#pragma unroll
    for (int dc = 0; dc < 4; ++dc)
#pragma unroll
      for (int r = 0; r < 8; ++r) o[dc][r] *= alpha[r];

    // C-layout -> A-layout transpose of P through LDS.
#pragma unroll
    for (int r = 0; r < 8; ++r) {
      myP[(prow + r) * 32 + pc] = (_Float16)s0[r];
      myP[(prow + r) * 32 + 16 + pc] = (_Float16)s1[r];
    }
    __syncthreads();
    const v16h pf = ldA_h(myP, 32, lane);
#pragma unroll
    for (int dc = 0; dc < 4; ++dc) o[dc] = WMMA(pf, vb[dc], o[dc]);
    __syncthreads();  // protect P tile before next iteration's stores
  }

  const int b = bh >> 4, h = bh & 15;
#pragma unroll
  for (int dc = 0; dc < 4; ++dc) {
#pragma unroll
    for (int r = 0; r < 8; ++r) {
      const int q = q0 + r + 8 * (lane >> 4);
      const int e = h * HDIM + dc * 16 + pc;
      Ah[((size_t)(b * SEQ) + q) * EMBED + e] = (_Float16)(o[dc][r] / lrow[r]);
    }
  }
}

// ---------------------------------------------------------------------------
extern "C" void kernel_launch(void* const* d_in, const int* in_sizes, int n_in,
                              void* d_out, int out_size, void* d_ws, size_t ws_size,
                              hipStream_t stream) {
  const float* query = (const float*)d_in[0];
  const float* key   = (const float*)d_in[1];
  const float* value = (const float*)d_in[2];
  const float* Wq = (const float*)d_in[3];
  const float* bq = (const float*)d_in[4];
  const float* Wk = (const float*)d_in[5];
  const float* bk = (const float*)d_in[6];
  const float* Wv = (const float*)d_in[7];
  const float* bv = (const float*)d_in[8];
  const float* Wo = (const float*)d_in[9];
  const float* bo = (const float*)d_in[10];

  const size_t me = (size_t)MTOT * EMBED;   // 8.39M elems
  const size_t we = (size_t)EMBED * EMBED;  // 1.05M elems

  _Float16* Qh  = (_Float16*)d_ws;          // [B,H,L,D]
  _Float16* Kh  = Qh + me;                  // [B,H,L,D]
  _Float16* Vt  = Kh + me;                  // [B,H,D,L]
  _Float16* Wqh = Vt + me;
  _Float16* Wkh = Wqh + we;
  _Float16* Wvh = Wkh + we;
  _Float16* Woh = Wvh + we;
  _Float16* Xb  = Woh + we;                 // reused activation buffer [M,1024]

  const int cvtThreads = 256;
  const int wg8 = (int)(we / 8), mg8 = (int)(me / 8);
  cvt_f32_f16<<<wg8 / cvtThreads, cvtThreads, 0, stream>>>(Wq, Wqh, wg8);
  cvt_f32_f16<<<wg8 / cvtThreads, cvtThreads, 0, stream>>>(Wk, Wkh, wg8);
  cvt_f32_f16<<<wg8 / cvtThreads, cvtThreads, 0, stream>>>(Wv, Wvh, wg8);
  cvt_f32_f16<<<wg8 / cvtThreads, cvtThreads, 0, stream>>>(Wo, Woh, wg8);

  dim3 ggrid(MTOT / 256, EMBED / 64);  // 32 x 16

  cvt_f32_f16<<<mg8 / cvtThreads, cvtThreads, 0, stream>>>(query, Xb, mg8);
  mha_gemm_wmma<<<ggrid, 256, 0, stream>>>(Xb, Wqh, bq, Qh, nullptr, 0);
  cvt_f32_f16<<<mg8 / cvtThreads, cvtThreads, 0, stream>>>(key, Xb, mg8);
  mha_gemm_wmma<<<ggrid, 256, 0, stream>>>(Xb, Wkh, bk, Kh, nullptr, 0);
  cvt_f32_f16<<<mg8 / cvtThreads, cvtThreads, 0, stream>>>(value, Xb, mg8);
  mha_gemm_wmma<<<ggrid, 256, 0, stream>>>(Xb, Wvh, bv, Vt, nullptr, 1);

  mha_attn_wmma<<<dim3(SEQ / 64, BATCH * NHEAD), 128, 0, stream>>>(Qh, Kh, Vt, Xb);

  mha_gemm_wmma<<<ggrid, 256, 0, stream>>>(Xb, Woh, bo, nullptr, (float*)d_out, 2);
}
